// UnsupervisedGAT_9242769622456
// MI455X (gfx1250) — matrix operand order repaired and checked by reading, not compiled, not run on hardware
//
#include <hip/hip_runtime.h>
#include <hip/hip_bf16.h>

// ---------------------------------------------------------------------------
// 2-layer GAT for MI455X (gfx1250, wave32).
//   GEMMs via v_wmma_f32_16x16x32_f16 (f16 operands, f32 accumulate).
//   Edge phase via guaranteed-hardware global_atomic_add_f32 (inline asm),
//   scattering into accumulators that are fully L2-resident (192 MB L2).
//   Softmax fused: rst = (sum_e w*feat[src]) / (sum_e w), w = exp(lrelu(el+er)).
// ---------------------------------------------------------------------------

#define N_NODES 100000
#define N_EDGES 1600000
#define IN_DIM  128
#define HID     64
#define HEADS   8
#define OUTF    8

typedef __attribute__((ext_vector_type(16))) _Float16 v16h;
typedef __attribute__((ext_vector_type(8)))  _Float16 v8h;
typedef __attribute__((ext_vector_type(8)))  float    v8f;

// Non-returning hardware FP atomic add (CDNA5 GLOBAL_ATOMIC_ADD_F32).
// Avoids any chance of LLVM expanding atomicrmw fadd into a CAS loop.
// Literal byte offset folds into the instruction's IOFFSET field so eight
// per-head accumulates share one 64-bit address.
template <int OFF>
__device__ __forceinline__ void atomic_add_f32(float* p, float v) {
    asm volatile("global_atomic_add_f32 %0, %1, off offset:%2"
                 :
                 : "v"(p), "v"(v), "i"(OFF)
                 : "memory");
}

// ---------------------------------------------------------------------------
// Transpose + convert weights: Wt[c][k] = (f16) W[k][c];  W is [din][64].
// ---------------------------------------------------------------------------
__global__ void gat_wconv_kernel(const float* __restrict__ W,
                                 _Float16* __restrict__ Wt, int din) {
    int t = blockIdx.x * blockDim.x + threadIdx.x;
    int total = 64 * din;
    if (t >= total) return;
    int c = t / din;
    int k = t - c * din;
    Wt[c * din + k] = (_Float16)W[k * 64 + c];
}

// ---------------------------------------------------------------------------
// GEMM: feat[N,64] = X[N,DIN] @ W[DIN,64], via WMMA.
// Block = 128 threads = 4 waves; each block does a 16-row slab, each wave a
// 16-column tile. A staged in LDS as f16; B read from pre-transposed Wt.
// ---------------------------------------------------------------------------
template <int DIN>
__global__ __launch_bounds__(128) void gat_gemm_kernel(
    const float* __restrict__ X, const _Float16* __restrict__ Wt,
    float* __restrict__ feat) {
    __shared__ _Float16 As[16 * DIN];

    const int row0 = blockIdx.x * 16;
    const int tid  = threadIdx.x;

    // Cooperative load + f32->f16 convert of the 16xDIN activation tile.
    for (int i = tid; i < 16 * DIN; i += 128) {
        As[i] = (_Float16)X[(size_t)row0 * DIN + i];
    }
    __syncthreads();

    const int wave = tid >> 5;
    const int lane = tid & 31;
    const int r    = lane & 15;   // A row within tile
    const int hsel = lane >> 4;   // high/low K-half selector
    const int col  = wave * 16 + (lane & 15);  // output column (B column)

    v8f acc = {};
#pragma unroll
    for (int kb = 0; kb < DIN; kb += 32) {
        // A frag (16-bit A 16x32 layout): lane holds row r, two runs of 8 K's:
        //   elems 0..7  -> K = kb + hsel*8 + j
        //   elems 8..15 -> K = kb + 16 + hsel*8 + (j-8)
        const v8h a_lo = *(const v8h*)(&As[r * DIN + kb + hsel * 8]);
        const v8h a_hi = *(const v8h*)(&As[r * DIN + kb + 16 + hsel * 8]);
        v16h a = __builtin_shufflevector(a_lo, a_hi,
                                         0, 1, 2, 3, 4, 5, 6, 7,
                                         8, 9, 10, 11, 12, 13, 14, 15);
        // B frag (16-bit B 32x16 layout): lane holds column `col`,
        //   elems 0..7 -> K = kb + hsel*8 + j, elems 8..15 -> K = kb + 16 + hsel*8 + (j-8).
        const v8h b_lo = *(const v8h*)(Wt + (size_t)col * DIN + kb + hsel * 8);
        const v8h b_hi = *(const v8h*)(Wt + (size_t)col * DIN + kb + 16 + hsel * 8);
        v16h b = __builtin_shufflevector(b_lo, b_hi,
                                         0, 1, 2, 3, 4, 5, 6, 7,
                                         8, 9, 10, 11, 12, 13, 14, 15);

        acc = __builtin_amdgcn_wmma_f32_16x16x32_f16(
            false, a, false, b, (short)0, acc, false, false);
    }

    // C layout: VGPR v, lanes 0-15 -> M=v, lanes 16-31 -> M=8+v; N = lane&15.
#pragma unroll
    for (int v = 0; v < 8; ++v) {
        int m = v + hsel * 8;
        feat[(size_t)(row0 + m) * 64 + col] = acc[v];
    }
}

// ---------------------------------------------------------------------------
// el/er projections: el[n,h] = sum_f feat[n,h,f]*al[h,f]  (and er with ar).
// One thread per (node, head).
// ---------------------------------------------------------------------------
__global__ void gat_eler_kernel(const float* __restrict__ feat,
                                const float* __restrict__ al,
                                const float* __restrict__ ar,
                                float* __restrict__ el,
                                float* __restrict__ er, int n_nodes) {
    int t = blockIdx.x * blockDim.x + threadIdx.x;
    int n = t >> 3;
    int h = t & 7;
    if (n >= n_nodes) return;
    const float* f = feat + (size_t)n * 64 + h * 8;
    const float* a = al + h * 8;
    const float* b = ar + h * 8;
    float sl = 0.f, sr = 0.f;
#pragma unroll
    for (int i = 0; i < 8; ++i) {
        float v = f[i];
        sl += v * a[i];
        sr += v * b[i];
    }
    el[n * 8 + h] = sl;
    er[n * 8 + h] = sr;
}

// ---------------------------------------------------------------------------
// Edge phase: 8 lanes per edge (one per head).
//   w = exp(lrelu(el[src,h] + er[dst,h], 0.2))
//   s[dst,h]       += w                      (1 hw atomic, L2-resident)
//   rst[dst,h,0:8] += w * feat[src,h,0:8]    (8 hw atomics, L2-resident)
// Softmax max-subtraction is algebraically cancelled (e is bounded ~|3|).
// ---------------------------------------------------------------------------
__global__ __launch_bounds__(256) void gat_edge_kernel(
    const int* __restrict__ src, const int* __restrict__ dst,
    const float* __restrict__ el, const float* __restrict__ er,
    const float* __restrict__ feat, float* __restrict__ s,
    float* __restrict__ rst, int n_edges) {
    int t = blockIdx.x * blockDim.x + threadIdx.x;
    int e = t >> 3;
    int h = t & 7;
    if (e >= n_edges) return;

    int sn = src[e];
    int dn = dst[e];

    float x = el[sn * 8 + h] + er[dn * 8 + h];
    x = (x >= 0.f) ? x : 0.2f * x;          // ATTN_SLOPE
    float w = __expf(x);

    atomic_add_f32<0>(&s[dn * 8 + h], w);

    const float4* f4 = (const float4*)(feat + (size_t)sn * 64 + h * 8);
    float4 fa = f4[0];
    float4 fb = f4[1];
    float* o = rst + (size_t)dn * 64 + h * 8;
    atomic_add_f32<0>(o, w * fa.x);
    atomic_add_f32<4>(o, w * fa.y);
    atomic_add_f32<8>(o, w * fa.z);
    atomic_add_f32<12>(o, w * fa.w);
    atomic_add_f32<16>(o, w * fb.x);
    atomic_add_f32<20>(o, w * fb.y);
    atomic_add_f32<24>(o, w * fb.z);
    atomic_add_f32<28>(o, w * fb.w);
}

// ---------------------------------------------------------------------------
// Finalize in place: rst[n,h,f] = rst[n,h,f]/s[n,h] + b[h,f]  (+ optional
// leaky-relu 0.01). Nodes with no incoming edges (s==0) get rst=b.
// ---------------------------------------------------------------------------
__global__ void gat_final_kernel(float* __restrict__ rst,
                                 const float* __restrict__ s,
                                 const float* __restrict__ b, int n_nodes,
                                 int apply_act) {
    int t = blockIdx.x * blockDim.x + threadIdx.x;
    int n = t >> 3;
    int h = t & 7;
    if (n >= n_nodes) return;
    float denom = s[n * 8 + h];
    float inv = (denom > 0.f) ? (1.0f / denom) : 0.f;
    float* p = rst + (size_t)n * 64 + h * 8;
    const float* bb = b + h * 8;
#pragma unroll
    for (int f = 0; f < 8; ++f) {
        float v = p[f] * inv + bb[f];
        if (apply_act) v = (v >= 0.f) ? v : 0.01f * v;   // ACT_SLOPE
        p[f] = v;
    }
}

// ---------------------------------------------------------------------------
// Host-side orchestration.
// ---------------------------------------------------------------------------
extern "C" void kernel_launch(void* const* d_in, const int* in_sizes, int n_in,
                              void* d_out, int out_size, void* d_ws,
                              size_t ws_size, hipStream_t stream) {
    const float* n_feat = (const float*)d_in[0];
    const int*   src    = (const int*)d_in[1];
    const int*   dst    = (const int*)d_in[2];
    const float* W0     = (const float*)d_in[3];
    const float* al0    = (const float*)d_in[4];
    const float* ar0    = (const float*)d_in[5];
    const float* b0     = (const float*)d_in[6];
    const float* W1     = (const float*)d_in[7];
    const float* al1    = (const float*)d_in[8];
    const float* ar1    = (const float*)d_in[9];
    const float* b1     = (const float*)d_in[10];

    char* ws = (char*)d_ws;
    const size_t FEAT_B = (size_t)N_NODES * 64 * sizeof(float);   // 25.6 MB
    const size_t EL_B   = (size_t)N_NODES * 8 * sizeof(float);    //  3.2 MB
    float*    feat = (float*)(ws);
    float*    rst0 = (float*)(ws + FEAT_B);
    float*    el   = (float*)(ws + 2 * FEAT_B);
    float*    er   = (float*)(ws + 2 * FEAT_B + EL_B);
    float*    sacc = (float*)(ws + 2 * FEAT_B + 2 * EL_B);
    _Float16* Wt0  = (_Float16*)(ws + 2 * FEAT_B + 3 * EL_B);
    _Float16* Wt1  = (_Float16*)(ws + 2 * FEAT_B + 3 * EL_B +
                                 (size_t)64 * IN_DIM * sizeof(_Float16));
    float* out = (float*)d_out;

    const int nh_blocks   = (N_NODES * 8 + 255) / 256;      // node*head grids
    const int edge_blocks = (N_EDGES * 8 + 255) / 256;      // edge*head grid
    const int gemm_blocks = N_NODES / 16;                   // 100000 % 16 == 0

    // ---- weight conversion (tiny) ----
    gat_wconv_kernel<<<(64 * IN_DIM + 255) / 256, 256, 0, stream>>>(W0, Wt0, IN_DIM);
    gat_wconv_kernel<<<(64 * HID + 255) / 256, 256, 0, stream>>>(W1, Wt1, HID);

    // ================= Layer 0 =================
    gat_gemm_kernel<IN_DIM><<<gemm_blocks, 128, 0, stream>>>(n_feat, Wt0, feat);
    gat_eler_kernel<<<nh_blocks, 256, 0, stream>>>(feat, al0, ar0, el, er, N_NODES);
    hipMemsetAsync(sacc, 0, EL_B, stream);
    hipMemsetAsync(rst0, 0, FEAT_B, stream);
    gat_edge_kernel<<<edge_blocks, 256, 0, stream>>>(src, dst, el, er, feat,
                                                     sacc, rst0, N_EDGES);
    gat_final_kernel<<<nh_blocks, 256, 0, stream>>>(rst0, sacc, b0, N_NODES, 1);

    // ================= Layer 1 =================
    gat_gemm_kernel<HID><<<gemm_blocks, 128, 0, stream>>>(rst0, Wt1, feat);
    gat_eler_kernel<<<nh_blocks, 256, 0, stream>>>(feat, al1, ar1, el, er, N_NODES);
    hipMemsetAsync(sacc, 0, EL_B, stream);
    hipMemsetAsync(out, 0, FEAT_B, stream);
    gat_edge_kernel<<<edge_blocks, 256, 0, stream>>>(src, dst, el, er, feat,
                                                     sacc, out, N_EDGES);
    gat_final_kernel<<<nh_blocks, 256, 0, stream>>>(out, sacc, b1, N_NODES, 0);
}
